// myGIN_4157528342729
// MI455X (gfx1250) — matrix operand order repaired and checked by reading, not compiled, not run on hardware
//
#include <hip/hip_runtime.h>

typedef __attribute__((ext_vector_type(16))) _Float16 v16h;
typedef __attribute__((ext_vector_type(8)))  float    v8f;

#define HID 64

// ---------------------------------------------------------------- elementwise
__global__ void __launch_bounds__(256) scale_init_k(const float* __restrict__ in,
                                                    const float* __restrict__ eps,
                                                    float* __restrict__ out, int n) {
    int i = blockIdx.x * 256 + threadIdx.x;
    if (i < n) out[i] = (1.0f + eps[0]) * in[i];
}

// ---------------------------------------------------------------- scatter add
__global__ void __launch_bounds__(256) scatter_add_k(const float* __restrict__ feat,
                                                     const long long* __restrict__ ei,
                                                     float* __restrict__ agg, long long E) {
    long long idx = (long long)blockIdx.x * 256 + threadIdx.x;
    if (idx >= E * 16) return;
    long long e = idx >> 4;
    int f4 = (int)(idx & 15) * 4;
    long long src = ei[e];
    long long dst = ei[E + e];
    const float4 v = *(const float4*)(feat + src * HID + f4);
    float* p = agg + dst * HID + f4;
    unsafeAtomicAdd(p + 0, v.x);
    unsafeAtomicAdd(p + 1, v.y);
    unsafeAtomicAdd(p + 2, v.z);
    unsafeAtomicAdd(p + 3, v.w);
}

// ---------------------------------------------------------------- WMMA GEMM
// out[N,64] = relu(in[N,64] @ W[64,64] + b), f16 WMMA with f32 accumulate.
// One wave handles a 16-row x 64-col strip (4 tiles of 16x16, K=64 -> 2 WMMAs/tile).
__global__ void __launch_bounds__(256) gemm_bias_relu_k(const float* __restrict__ in,
                                                        const float* __restrict__ W,
                                                        const float* __restrict__ bias,
                                                        float* __restrict__ out, int nRows) {
    __shared__ __align__(32) _Float16 Wt[HID * HID];  // Wt[n*64 + k]  (transposed)
    __shared__ float bLds[HID];

    int tid = threadIdx.x;
    for (int i = tid; i < HID * HID; i += 256) {
        int k = i >> 6, n = i & 63;
        Wt[n * HID + k] = (_Float16)W[k * HID + n];
    }
    if (tid < HID) bLds[tid] = bias[tid];
    __syncthreads();

    int wave = tid >> 5;
    int lane = tid & 31;
    int m0 = (blockIdx.x * 8 + wave) * 16;
    if (m0 >= nRows) return;            // uniform per-wave exit (EXEC stays all-1 for WMMA)

    const bool full = (m0 + 16 <= nRows);
    int half = lane >> 4;               // which 16-lane half
    int l15  = lane & 15;
    int row  = m0 + l15;
    if (row >= nRows) row = nRows - 1;  // only matters on (rare) partial tile
    const float* rowp = in + (long long)row * HID;
    int klo = half * 8;

    // A fragments: 16-bit A 16x32 layout.  elems 0..7 -> K = k0+klo+e,
    // elems 8..15 -> K = k0+16+klo+(e-8)
    v16h a[2];
#pragma unroll
    for (int c = 0; c < 2; ++c) {
        int k0 = c * 32;
        const float4 f0 = *(const float4*)(rowp + k0 + klo);
        const float4 f1 = *(const float4*)(rowp + k0 + klo + 4);
        const float4 f2 = *(const float4*)(rowp + k0 + 16 + klo);
        const float4 f3 = *(const float4*)(rowp + k0 + 16 + klo + 4);
        a[c][0]  = (_Float16)f0.x; a[c][1]  = (_Float16)f0.y;
        a[c][2]  = (_Float16)f0.z; a[c][3]  = (_Float16)f0.w;
        a[c][4]  = (_Float16)f1.x; a[c][5]  = (_Float16)f1.y;
        a[c][6]  = (_Float16)f1.z; a[c][7]  = (_Float16)f1.w;
        a[c][8]  = (_Float16)f2.x; a[c][9]  = (_Float16)f2.y;
        a[c][10] = (_Float16)f2.z; a[c][11] = (_Float16)f2.w;
        a[c][12] = (_Float16)f3.x; a[c][13] = (_Float16)f3.y;
        a[c][14] = (_Float16)f3.z; a[c][15] = (_Float16)f3.w;
    }

#pragma unroll
    for (int t = 0; t < 4; ++t) {
        int n0 = t * 16;
        int col = n0 + l15;
        v8f acc = {};
#pragma unroll
        for (int c = 0; c < 2; ++c) {
            int k0 = c * 32;
            // B fragment: lane col = n&15, elems 0..7 -> K = k0+klo+e, elems 8..15 -> K = k0+16+klo+(e-8)
            typedef __attribute__((ext_vector_type(8))) _Float16 v8h_;
            const v8h_ blo = *(const v8h_*)(&Wt[col * HID + k0 + klo]);
            const v8h_ bhi = *(const v8h_*)(&Wt[col * HID + k0 + 16 + klo]);
            const v16h b = __builtin_shufflevector(blo, bhi, 0,1,2,3,4,5,6,7,8,9,10,11,12,13,14,15);
            acc = __builtin_amdgcn_wmma_f32_16x16x32_f16(
                false, a[c], false, b, (short)0, acc, false, false);
        }
        float bv = bLds[col];
        // C layout: VGPR v -> M = v + 8*half, this lane's column = col
        float* op = out + (long long)(m0 + half * 8) * HID + col;
        if (full) {
#pragma unroll
            for (int v = 0; v < 8; ++v) {
                float x = acc[v] + bv;
                op[v * HID] = x > 0.0f ? x : 0.0f;   // branch-free, imm-offset stores
            }
        } else {
#pragma unroll
            for (int v = 0; v < 8; ++v) {
                int r = m0 + half * 8 + v;
                if (r < nRows) {
                    float x = acc[v] + bv;
                    op[v * HID] = x > 0.0f ? x : 0.0f;
                }
            }
        }
    }
}

// ---------------------------------------------------------------- batch norm
__global__ void zero_stats_k(float* stats) {
    int i = threadIdx.x;
    if (i < 128) stats[i] = 0.0f;
}

__global__ void __launch_bounds__(256) bn_stats_k(const float* __restrict__ h,
                                                  float* __restrict__ stats, int nRows) {
    __shared__ float sh[2][4][HID];
    int tid = threadIdx.x;
    int f = tid & 63, sub = tid >> 6;
    int r0 = blockIdx.x * 512;
    int rEnd = r0 + 512; if (rEnd > nRows) rEnd = nRows;
    float s = 0.0f, ss = 0.0f;
    for (int r = r0 + sub; r < rEnd; r += 4) {
        float v = h[(long long)r * HID + f];
        s += v; ss += v * v;
    }
    sh[0][sub][f] = s; sh[1][sub][f] = ss;
    __syncthreads();
    if (sub == 0) {
        float ts  = sh[0][0][f] + sh[0][1][f] + sh[0][2][f] + sh[0][3][f];
        float tss = sh[1][0][f] + sh[1][1][f] + sh[1][2][f] + sh[1][3][f];
        unsafeAtomicAdd(&stats[f], ts);
        unsafeAtomicAdd(&stats[HID + f], tss);
    }
}

__global__ void __launch_bounds__(256) bn_apply_k(float* __restrict__ h,
                                                  const float* __restrict__ stats,
                                                  const float* __restrict__ g,
                                                  const float* __restrict__ bt,
                                                  int n, float invN) {
    int i = blockIdx.x * 256 + threadIdx.x;
    if (i >= n) return;
    int f = i & 63;
    float mean = stats[f] * invN;
    float var  = stats[HID + f] * invN - mean * mean;   // biased, matches reference
    float inv  = rsqrtf(var + 1e-5f);
    h[i] = (h[i] - mean) * inv * g[f] + bt[f];
}

// ---------------------------------------------------------------- log_softmax
__global__ void __launch_bounds__(256) log_softmax_k(const float* __restrict__ h,
                                                     float* __restrict__ out, int nRows) {
    int row  = blockIdx.x * 8 + (threadIdx.x >> 5);
    int lane = threadIdx.x & 31;
    if (row >= nRows) return;
    float v0 = h[(long long)row * HID + lane];
    float v1 = h[(long long)row * HID + lane + 32];
    float m = fmaxf(v0, v1);
    for (int o = 16; o > 0; o >>= 1) m = fmaxf(m, __shfl_xor(m, o, 32));
    float s = __expf(v0 - m) + __expf(v1 - m);
    for (int o = 16; o > 0; o >>= 1) s += __shfl_xor(s, o, 32);
    float lse = m + __logf(s);
    out[(long long)row * HID + lane]      = v0 - lse;
    out[(long long)row * HID + lane + 32] = v1 - lse;
}

// ---------------------------------------------------------------- launch
extern "C" void kernel_launch(void* const* d_in, const int* in_sizes, int n_in,
                              void* d_out, int out_size, void* d_ws, size_t ws_size,
                              hipStream_t stream) {
    const float*     x    = (const float*)d_in[0];
    const long long* ei   = (const long long*)d_in[1];
    const float*     W1   = (const float*)d_in[2];
    const float*     b1   = (const float*)d_in[3];
    const float*     W2   = (const float*)d_in[4];
    const float*     b2   = (const float*)d_in[5];
    const float*     g1   = (const float*)d_in[6];
    const float*     bt1  = (const float*)d_in[7];
    const float*     eps1 = (const float*)d_in[8];
    const float*     W3   = (const float*)d_in[9];
    const float*     b3   = (const float*)d_in[10];
    const float*     W4   = (const float*)d_in[11];
    const float*     b4   = (const float*)d_in[12];
    const float*     g2   = (const float*)d_in[13];
    const float*     bt2  = (const float*)d_in[14];
    const float*     eps2 = (const float*)d_in[15];

    const int       N  = in_sizes[0] / HID;
    const long long E  = in_sizes[1] / 2;
    const int nElem    = N * HID;

    float* A     = (float*)d_ws;                  // N*64
    float* B     = A + (size_t)nElem;             // N*64
    float* stats = B + (size_t)nElem;             // 128
    float* outH  = (float*)d_out;                 // layer-2 work buffer == final h
    float* outL  = outH + (size_t)nElem;          // final log_softmax

    const dim3 blk(256);
    const int gElem  = (nElem + 255) / 256;
    const int gScat  = (int)((E * 16 + 255) / 256);
    const int rowT   = (N + 15) / 16;
    const int gGemm  = (rowT + 7) / 8;
    const int gStats = (N + 511) / 512;
    const int gRows  = (N + 7) / 8;
    const float invN = 1.0f / (float)N;

    // ---- layer 1 ----
    scale_init_k<<<gElem, blk, 0, stream>>>(x, eps1, A, nElem);
    scatter_add_k<<<gScat, blk, 0, stream>>>(x, ei, A, E);
    gemm_bias_relu_k<<<gGemm, blk, 0, stream>>>(A, W1, b1, B, N);
    gemm_bias_relu_k<<<gGemm, blk, 0, stream>>>(B, W2, b2, A, N);
    zero_stats_k<<<1, 128, 0, stream>>>(stats);
    bn_stats_k<<<gStats, blk, 0, stream>>>(A, stats, N);
    bn_apply_k<<<gElem, blk, 0, stream>>>(A, stats, g1, bt1, nElem, invN);

    // ---- layer 2 (work buffer = first half of d_out; h2 lands in place) ----
    scale_init_k<<<gElem, blk, 0, stream>>>(A, eps2, outH, nElem);
    scatter_add_k<<<gScat, blk, 0, stream>>>(A, ei, outH, E);
    gemm_bias_relu_k<<<gGemm, blk, 0, stream>>>(outH, W3, b3, B, N);
    gemm_bias_relu_k<<<gGemm, blk, 0, stream>>>(B, W4, b4, outH, N);
    zero_stats_k<<<1, 128, 0, stream>>>(stats);
    bn_stats_k<<<gStats, blk, 0, stream>>>(outH, stats, N);
    bn_apply_k<<<gElem, blk, 0, stream>>>(outH, stats, g2, bt2, nElem, invN);

    // ---- outputs ----
    log_softmax_k<<<gRows, blk, 0, stream>>>(outH, outL, N);
}
